// CrossDomain_PSA_4337916969561
// MI455X (gfx1250) — matrix operand order repaired and checked by reading, not compiled, not run on hardware
//
#include <hip/hip_runtime.h>
#include <hip/hip_bf16.h>

// ---------------------------------------------------------------------------
// Types for CDNA5 WMMA
// ---------------------------------------------------------------------------
typedef __attribute__((ext_vector_type(16))) __bf16   v16bf;
typedef __attribute__((ext_vector_type(8)))  float    v8f;
typedef __attribute__((ext_vector_type(4)))  unsigned u32x4;

#define PI_F 3.14159265358979323846f
#define BNS  0.99999500003749969f   // 1/sqrt(1 + 1e-5)

__device__ __forceinline__ unsigned f2bfu(float f) {
    union { float f; unsigned u; } x; x.f = f;
    unsigned r = x.u + 0x7FFFu + ((x.u >> 16) & 1u);
    return r >> 16;
}
// view into channel block b of y: image img = (batch*128 + c), 2nd half channels
__device__ __forceinline__ const float* bch_ptr(const float* y, int img) {
    return y + (long)(img >> 7) * 262144 + 131072 + (long)(img & 127) * 1024;
}

// ---------------------------------------------------------------------------
// Branchless element fetchers
// A modes: 0 row-major  1 transposed (A[k*lda+m])  2 tconv-weight gather
// B modes: 0 row-major  1 transposed (B[n*ldb+k])  2 im2col(3x3,pad1) NCHW
// ---------------------------------------------------------------------------
template<int AM>
__device__ __forceinline__ float fetchA(const float* __restrict__ A, int m, int k,
                                        int lda, int aDim1)
{
    if (AM == 0) return A[(long)m * lda + k];
    if (AM == 1) return A[(long)k * lda + m];
    int ci = k / 9, rr = k - ci * 9;
    int ky = rr / 3, kx = rr - ky * 3;
    return A[((long)(ci * aDim1 + m)) * 9 + (2 - ky) * 3 + (2 - kx)];
}

template<int BM>
__device__ __forceinline__ float fetchB(const float* __restrict__ B, int k, int n,
                                        int ldb, int bH, int bW)
{
    if (BM == 0) return B[(long)k * ldb + n];
    if (BM == 1) return B[(long)n * ldb + k];
    int ci = k / 9, rr = k - ci * 9;
    int ky = rr / 3 - 1, kx = rr - (rr / 3) * 3 - 1;
    int oy = n / bW, ox = n - oy * bW;
    int iy = oy + ky, ix = ox + kx;
    bool ok = (iy >= 0) & (iy < bH) & (ix >= 0) & (ix < bW);
    int iyc = iy < 0 ? 0 : (iy >= bH ? bH - 1 : iy);
    int ixc = ix < 0 ? 0 : (ix >= bW ? bW - 1 : ix);
    float v = B[(long)ci * bH * bW + (long)iyc * bW + ixc];
    return ok ? v : 0.f;
}

// ---------------------------------------------------------------------------
// Tile staging: 64(m) x 32(k) for A, 32(k) x 64(n) for B (stored transposed).
// Each of 128 threads handles 8 bf16-pairs (packed u32) per tile.
// ---------------------------------------------------------------------------
template<int AM>
__device__ __forceinline__ void loadTileA(const float* __restrict__ A, int bm, int k0,
                                          int M, int K, int lda, int aDim1,
                                          unsigned* ra, int tid)
{
    #pragma unroll
    for (int e = 0; e < 8; ++e) {
        int p  = tid + 128 * e;          // pair index 0..1023
        int r  = p >> 4, c2 = p & 15;    // row 0..63, k-pair 0..15
        int m  = bm + r;
        bool mv = (m < M);
        int mm = mv ? m : (M - 1);
        int k1 = k0 + 2 * c2, k2 = k1 + 1;
        float v0 = fetchA<AM>(A, mm, (k1 < K ? k1 : K - 1), lda, aDim1);
        float v1 = fetchA<AM>(A, mm, (k2 < K ? k2 : K - 1), lda, aDim1);
        v0 = (mv && k1 < K) ? v0 : 0.f;
        v1 = (mv && k2 < K) ? v1 : 0.f;
        ra[e] = f2bfu(v0) | (f2bfu(v1) << 16);
    }
}

template<int BM>
__device__ __forceinline__ void loadTileB(const float* __restrict__ B, int bn, int k0,
                                          int N, int K, int ldb, int bH, int bW,
                                          unsigned* rb, int tid)
{
    #pragma unroll
    for (int e = 0; e < 8; ++e) {
        int p  = tid + 128 * e;
        int nl = p & 63, k2i = p >> 6;   // n 0..63 (fast for coalescing), k-pair 0..15
        int n  = bn + nl;
        bool nv = (n < N);
        int nn = nv ? n : (N - 1);
        int k1 = k0 + 2 * k2i, k2 = k1 + 1;
        float v0 = fetchB<BM>(B, (k1 < K ? k1 : K - 1), nn, ldb, bH, bW);
        float v1 = fetchB<BM>(B, (k2 < K ? k2 : K - 1), nn, ldb, bH, bW);
        v0 = (nv && k1 < K) ? v0 : 0.f;
        v1 = (nv && k2 < K) ? v1 : 0.f;
        rb[e] = f2bfu(v0) | (f2bfu(v1) << 16);
        if (BM == 0 && k0 + 64 < K)      // gfx1250 prefetch two tiles ahead
            __builtin_prefetch(&B[(long)(k1 + 64) * ldb + nn], 0, 1);
    }
}

__device__ __forceinline__ void storeTileA(unsigned short (&T)[64][40],
                                           const unsigned* ra, int tid)
{
    #pragma unroll
    for (int e = 0; e < 8; ++e) {
        int p = tid + 128 * e;
        int r = p >> 4, c2 = p & 15;
        *(unsigned*)&T[r][2 * c2] = ra[e];
    }
}
__device__ __forceinline__ void storeTileB(unsigned short (&T)[64][40],
                                           const unsigned* rb, int tid)
{
    #pragma unroll
    for (int e = 0; e < 8; ++e) {
        int p = tid + 128 * e;
        int nl = p & 63, k2i = p >> 6;
        *(unsigned*)&T[nl][2 * k2i] = rb[e];
    }
}

// Fragment loads: two ds_load_b128 each (rows padded to 40 halfwords = 80B).
__device__ __forceinline__ v16bf ldfragA(const unsigned short (&T)[64][40], int row, int hi)
{
    union { u32x4 q[2]; v16bf v; } u;
    u.q[0] = *(const u32x4*)&T[row][hi * 8];
    u.q[1] = *(const u32x4*)&T[row][16 + hi * 8];
    return u.v;
}
__device__ __forceinline__ v16bf ldfragB(const unsigned short (&T)[64][40], int col, int hi)
{
    union { u32x4 q[2]; v16bf v; } u;
    u.q[0] = *(const u32x4*)&T[col][hi * 8];
    u.q[1] = *(const u32x4*)&T[col][16 + hi * 8];
    return u.v;
}

// ---------------------------------------------------------------------------
// Implicit-GEMM WMMA kernel: 64x64 block tile, 4 waves, 32x32 per wave
// (2x2 v_wmma_f32_16x16x32_bf16 per K-step), double-buffered LDS pipeline.
//   C[m,n] = act( scale[m] * sum_k A[m,k]*B[k,n] + bias )
// ---------------------------------------------------------------------------
template<int AM, int BM>
__global__ void gemm_wmma(const float* __restrict__ A, const float* __restrict__ B,
                          float* __restrict__ C,
                          int M, int N, int K, int lda, int ldb, int ldc,
                          long sA, long sB, long sC,
                          const float* __restrict__ gamma, const float* __restrict__ beta,
                          int biasMode, int act,
                          int bH, int bW, int aDim1)
{
    __shared__ unsigned short As[2][64][40];
    __shared__ unsigned short Bt[2][64][40];

    const int tid  = threadIdx.x;          // 0..127 (4 waves)
    const int lane = tid & 31;
    const int wave = tid >> 5;
    const int wrow = (wave >> 1) * 32;
    const int wcol = (wave & 1) * 32;
    const int bm = blockIdx.y * 64;
    const int bn = blockIdx.x * 64;
    A += blockIdx.z * sA; B += blockIdx.z * sB; C += blockIdx.z * sC;

    unsigned ra[8], rb[8];
    loadTileA<AM>(A, bm, 0, M, K, lda, aDim1, ra, tid);
    loadTileB<BM>(B, bn, 0, N, K, ldb, bH, bW, rb, tid);
    storeTileA(As[0], ra, tid);
    storeTileB(Bt[0], rb, tid);
    __syncthreads();

    v8f acc00 = {0,0,0,0,0,0,0,0}, acc01 = acc00, acc10 = acc00, acc11 = acc00;
    int cur = 0;
    for (int k0 = 0; k0 < K; k0 += 32) {
        const bool more = (k0 + 32 < K);
        if (more) {   // issue next tile's global loads before the math
            loadTileA<AM>(A, bm, k0 + 32, M, K, lda, aDim1, ra, tid);
            loadTileB<BM>(B, bn, k0 + 32, N, K, ldb, bH, bW, rb, tid);
        }
        const int r15 = lane & 15, hi = lane >> 4;
        v16bf a0 = ldfragA(As[cur], wrow + r15,      hi);
        v16bf a1 = ldfragA(As[cur], wrow + 16 + r15, hi);
        v16bf b0 = ldfragB(Bt[cur], wcol + r15,      hi);
        v16bf b1 = ldfragB(Bt[cur], wcol + 16 + r15, hi);
        acc00 = __builtin_amdgcn_wmma_f32_16x16x32_bf16(false, a0, false, b0, (short)0, acc00, false, false);
        acc01 = __builtin_amdgcn_wmma_f32_16x16x32_bf16(false, a0, false, b1, (short)0, acc01, false, false);
        acc10 = __builtin_amdgcn_wmma_f32_16x16x32_bf16(false, a1, false, b0, (short)0, acc10, false, false);
        acc11 = __builtin_amdgcn_wmma_f32_16x16x32_bf16(false, a1, false, b1, (short)0, acc11, false, false);
        if (more) {
            storeTileA(As[cur ^ 1], ra, tid);
            storeTileB(Bt[cur ^ 1], rb, tid);
        }
        __syncthreads();
        cur ^= 1;
    }

    // epilogue: bn-scale, bias, activation
    auto store16 = [&](const v8f& a, int mbase, int col) {
        #pragma unroll
        for (int r = 0; r < 8; ++r) {
            int m = mbase + r;
            if (m < M && col < N) {
                float v = a[r];
                if (gamma)              v *= gamma[m] * BNS;
                if (biasMode == 1)      v += beta[m];
                else if (biasMode == 2) v += beta[col];
                if (act == 1)           v = fmaxf(v, 0.f);
                else if (act == 2)      v = v / (1.f + __expf(-v));   // silu
                else if (act == 3)      v = 1.f / (1.f + __expf(-v)); // sigmoid
                C[(long)m * ldc + col] = v;
            }
        }
    };
    const int r15 = lane & 15, hi8 = (lane >> 4) * 8;
    store16(acc00, bm + wrow + hi8,      bn + wcol + r15);
    store16(acc01, bm + wrow + hi8,      bn + wcol + 16 + r15);
    store16(acc10, bm + wrow + 16 + hi8, bn + wcol + r15);
    store16(acc11, bm + wrow + 16 + hi8, bn + wcol + 16 + r15);
}

// ---------------------------------------------------------------------------
// Irregular-compute scalar kernels
// ---------------------------------------------------------------------------
__global__ void k_dft_rows(const float* __restrict__ y, float* __restrict__ tr,
                           float* __restrict__ ti)
{
    int idx = blockIdx.x * blockDim.x + threadIdx.x;
    if (idx >= 512 * 1024) return;
    int img = idx >> 10, rem = idx & 1023;
    int u = rem >> 5, j = rem & 31;
    const float* x = bch_ptr(y, img);
    float sr = 0.f, si = 0.f;
    for (int i = 0; i < 32; ++i) {
        float ang = -2.f * PI_F * (float)(u * i) / 32.f;
        float v = x[i * 32 + j];
        sr += v * __cosf(ang);
        si += v * __sinf(ang);
    }
    tr[idx] = sr; ti[idx] = si;
}

__global__ void k_dft_cols(const float* __restrict__ tr, const float* __restrict__ ti,
                           float* __restrict__ freq)
{
    int idx = blockIdx.x * blockDim.x + threadIdx.x;
    if (idx >= 512 * 1024) return;
    int img = idx >> 10, rem = idx & 1023;
    int u = rem >> 5, v = rem & 31;
    const float* r  = tr + (long)img * 1024;
    const float* im = ti + (long)img * 1024;
    float sr = 0.f, si = 0.f;
    for (int j = 0; j < 32; ++j) {
        float ang = -2.f * PI_F * (float)(v * j) / 32.f;
        float c = __cosf(ang), s = __sinf(ang);
        float a = r[u * 32 + j], b = im[u * 32 + j];
        sr += a * c - b * s;
        si += a * s + b * c;
    }
    int su = (u + 16) & 31, sv = (v + 16) & 31;
    freq[(long)img * 1024 + su * 32 + sv] = sqrtf(sr * sr + si * si);
}

__global__ void k_pool(const float* __restrict__ y, float* __restrict__ yl)
{
    int idx = blockIdx.x * blockDim.x + threadIdx.x;
    if (idx >= 512 * 256) return;
    int img = idx >> 8, rem = idx & 255;
    int p = rem >> 4, q = rem & 15;
    const float* x = bch_ptr(y, img);
    float s = x[(2*p)*32 + 2*q] + x[(2*p)*32 + 2*q+1]
            + x[(2*p+1)*32 + 2*q] + x[(2*p+1)*32 + 2*q+1];
    yl[idx] = 0.5f * s;
}

__global__ void k_radon(const float* __restrict__ y, float* __restrict__ sino)
{
    int idx = blockIdx.x * blockDim.x + threadIdx.x;
    if (idx >= 512 * 180 * 32) return;
    int img = idx / 5760, rem = idx % 5760;
    int t = rem / 32, j = rem & 31;
    const float* im = bch_ptr(y, img);
    float th = (float)t * (PI_F / 180.f);
    float cs = __cosf(th), sn = __sinf(th);
    float ctr = 15.5f, sum = 0.f;
    float dj = (float)j - ctr;
    for (int i = 0; i < 32; ++i) {
        float di = (float)i - ctr;
        float ys = ctr + di * cs - dj * sn;
        float xs = ctr + di * sn + dj * cs;
        int y0 = (int)floorf(ys), x0 = (int)floorf(xs);
        float fy = ys - (float)y0, fx = xs - (float)x0;
        float v = 0.f;
        if (y0   >= 0 && y0   < 32 && x0   >= 0 && x0   < 32) v += (1.f-fy)*(1.f-fx)*im[y0*32+x0];
        if (y0   >= 0 && y0   < 32 && x0+1 >= 0 && x0+1 < 32) v += (1.f-fy)*fx      *im[y0*32+x0+1];
        if (y0+1 >= 0 && y0+1 < 32 && x0   >= 0 && x0   < 32) v += fy*(1.f-fx)      *im[(y0+1)*32+x0];
        if (y0+1 >= 0 && y0+1 < 32 && x0+1 >= 0 && x0+1 < 32) v += fy*fx            *im[(y0+1)*32+x0+1];
        sum += v;
    }
    sino[idx] = sum;
}

// mode 0: sqrt(sum x^2)   mode 1: mean
__global__ void k_rowreduce(const float* __restrict__ src, float* __restrict__ out,
                            int HW, int mode)
{
    int row = blockIdx.x;
    const float* p = src + (long)row * HW;
    __shared__ float sh[256];
    float s = 0.f;
    for (int i = threadIdx.x; i < HW; i += 256) {
        float v = p[i];
        s += mode ? v : v * v;
    }
    sh[threadIdx.x] = s; __syncthreads();
    for (int st = 128; st > 0; st >>= 1) {
        if (threadIdx.x < st) sh[threadIdx.x] += sh[threadIdx.x + st];
        __syncthreads();
    }
    if (threadIdx.x == 0) out[row] = mode ? sh[0] / (float)HW : sqrtf(sh[0]);
}

__device__ float ot4(const float* X, const float* Y)
{
    float Cm[4][4];
    for (int i = 0; i < 4; ++i)
        for (int j = 0; j < 4; ++j) {
            float s = 0.f;
            for (int d = 0; d < 128; ++d) {
                float df = X[i * 128 + d] - Y[j * 128 + d];
                s += df * df;
            }
            Cm[i][j] = 0.5f * s;
        }
    const float eps = 0.0025f;            // 0.05^2
    const float lg  = -logf(4.f);
    float f[4] = {0,0,0,0}, g[4] = {0,0,0,0};
    for (int it = 0; it < 200; ++it) {
        float nf[4];
        for (int i = 0; i < 4; ++i) {
            float mx = -3.4e38f, t[4];
            for (int j = 0; j < 4; ++j) {
                t[j] = lg + (g[j] - Cm[i][j]) / eps;
                mx = fmaxf(mx, t[j]);
            }
            float s = 0.f;
            for (int j = 0; j < 4; ++j) s += __expf(t[j] - mx);
            nf[i] = -eps * (mx + logf(s));
        }
        for (int i = 0; i < 4; ++i) f[i] = nf[i];
        for (int j = 0; j < 4; ++j) {
            float mx = -3.4e38f, t[4];
            for (int i = 0; i < 4; ++i) {
                t[i] = lg + (f[i] - Cm[i][j]) / eps;
                mx = fmaxf(mx, t[i]);
            }
            float s = 0.f;
            for (int i = 0; i < 4; ++i) s += __expf(t[i] - mx);
            g[j] = -eps * (mx + logf(s));
        }
    }
    return 0.25f * (f[0]+f[1]+f[2]+f[3]) + 0.25f * (g[0]+g[1]+g[2]+g[3]);
}

__global__ void k_sinkhorn(const float* ef, const float* ew, const float* er, float* wout)
{
    if (threadIdx.x != 0 || blockIdx.x != 0) return;
    float d_fw = ot4(ef, ew) - 0.5f * ot4(ef, ef) - 0.5f * ot4(ew, ew);
    float d_fr = ot4(ef, er) - 0.5f * ot4(ef, ef) - 0.5f * ot4(er, er);
    float wfw = __expf(-d_fw), wfr = __expf(-d_fr);
    float tot = wfw + wfr;
    wout[0] = wfw / tot;   // fq == wv
    wout[1] = wfr / tot;   // rd
}

__global__ void k_scalevec(float* __restrict__ zs, const float* __restrict__ e,
                           const float* __restrict__ w, int widx)
{
    int i = blockIdx.x * blockDim.x + threadIdx.x;
    if (i < 512) zs[i] = e[i] * w[widx];
}

__global__ void k_dw_pe(const float* __restrict__ qkv, const float* __restrict__ w,
                        const float* __restrict__ g, const float* __restrict__ b,
                        float* __restrict__ out)
{
    int idx = blockIdx.x * blockDim.x + threadIdx.x;
    if (idx >= 524288) return;
    int bb = idx >> 17, c = (idx >> 10) & 127, p = idx & 1023;
    int yy = p >> 5, xx = p & 31;
    const float* v = qkv + (long)bb * 262144 + (long)(c >> 6) * 131072 + 65536 + (long)(c & 63) * 1024;
    float s = 0.f;
    for (int ky = 0; ky < 3; ++ky)
        for (int kx = 0; kx < 3; ++kx) {
            int iy = yy + ky - 1, ix = xx + kx - 1;
            if (iy >= 0 && iy < 32 && ix >= 0 && ix < 32)
                s += w[c * 9 + ky * 3 + kx] * v[iy * 32 + ix];
        }
    out[idx] = s * g[c] * BNS + b[c];
}

__global__ void k_softmax(float* __restrict__ S, float scale)
{
    long row = blockIdx.x;
    float* p = S + row * 1024;
    __shared__ float sh[256];
    float m = -3.4e38f;
    for (int i = threadIdx.x; i < 1024; i += 256) m = fmaxf(m, p[i] * scale);
    sh[threadIdx.x] = m; __syncthreads();
    for (int st = 128; st > 0; st >>= 1) {
        if (threadIdx.x < st) sh[threadIdx.x] = fmaxf(sh[threadIdx.x], sh[threadIdx.x + st]);
        __syncthreads();
    }
    float mx = sh[0]; __syncthreads();
    float s = 0.f;
    for (int i = threadIdx.x; i < 1024; i += 256) {
        float e = __expf(p[i] * scale - mx);
        p[i] = e; s += e;
    }
    sh[threadIdx.x] = s; __syncthreads();
    for (int st = 128; st > 0; st >>= 1) {
        if (threadIdx.x < st) sh[threadIdx.x] += sh[threadIdx.x + st];
        __syncthreads();
    }
    float inv = 1.f / sh[0];
    for (int i = threadIdx.x; i < 1024; i += 256) p[i] *= inv;
}

__global__ void k_comb(const float* __restrict__ y,
                       const float* __restrict__ df, const float* __restrict__ dw,
                       const float* __restrict__ dr,
                       const float* __restrict__ spf, const float* __restrict__ spw,
                       const float* __restrict__ spr,
                       float* __restrict__ comb)
{
    int idx = blockIdx.x * blockDim.x + threadIdx.x;
    if (idx >= 4 * 384 * 1024) return;
    int bb = idx / 393216, rem = idx % 393216;
    int ch = rem >> 10, p = rem & 1023;
    int t = ch >> 7, c = ch & 127;
    float bv = bch_ptr(y, bb * 128 + c)[p];
    const float* dec = (t == 0) ? df : (t == 1) ? dw : dr;
    const float* sp  = (t == 0) ? spf : (t == 1) ? spw : spr;
    comb[idx] = bv * dec[(long)bb * 131072 + c * 1024 + p] * sp[bb * 128 + c];
}

__global__ void k_bmul(const float* __restrict__ y, const float* __restrict__ cd,
                       float* __restrict__ out)
{
    int idx = blockIdx.x * blockDim.x + threadIdx.x;
    if (idx >= 524288) return;
    int bb = idx >> 17, r = idx & 131071;
    out[idx] = y[(long)bb * 262144 + 131072 + r] * cd[idx];
}

__global__ void k_add(float* __restrict__ dst, const float* __restrict__ a,
                      const float* __restrict__ b, int n)
{
    int i = blockIdx.x * blockDim.x + threadIdx.x;
    if (i < n) dst[i] = a[i] + b[i];
}

__global__ void k_cat(const float* __restrict__ y, const float* __restrict__ b4,
                      float* __restrict__ cat)
{
    int idx = blockIdx.x * blockDim.x + threadIdx.x;
    if (idx >= 1048576) return;
    int bb = idx >> 18, rem = idx & 262143;
    int ch = rem >> 10, p = rem & 1023;
    cat[idx] = (ch < 128) ? y[(long)bb * 262144 + ch * 1024 + p]
                          : b4[(long)bb * 131072 + (ch - 128) * 1024 + p];
}

// ---------------------------------------------------------------------------
// Host-side GEMM dispatch (64x64 block tiles)
// ---------------------------------------------------------------------------
template<int AM, int BM>
static inline void gemm(hipStream_t st, const float* A, const float* B, float* C,
                        int M, int N, int K, int lda, int ldb, int ldc,
                        long sA, long sB, long sC,
                        const float* g, const float* bt, int biasMode, int act,
                        int batches, int bH = 1, int bW = 1, int aDim1 = 0)
{
    dim3 gr((N + 63) / 64, (M + 63) / 64, batches);
    gemm_wmma<AM, BM><<<gr, dim3(128), 0, st>>>(A, B, C, M, N, K, lda, ldb, ldc,
                                                sA, sB, sC, g, bt, biasMode, act,
                                                bH, bW, aDim1);
}

// ---------------------------------------------------------------------------
extern "C" void kernel_launch(void* const* d_in, const int* in_sizes, int n_in,
                              void* d_out, int out_size, void* d_ws, size_t ws_size,
                              hipStream_t stream)
{
    (void)in_sizes; (void)n_in; (void)out_size; (void)ws_size;
    auto P = [&](int i) { return (const float*)d_in[i]; };
    const float* X = P(0);

    // workspace bump allocator (deterministic)
    float* W = (float*)d_ws;
    size_t off = 0;
    auto alloc = [&](size_t n) { float* p = W + off; off += n; return p; };

    float* y      = alloc(1048576);           // cv1 output [4][256][1024]
    float* tr     = alloc(524288);
    float* ti     = alloc(524288);
    float* freq   = alloc(524288);
    float* yl     = alloc(131072);
    float* sino   = alloc(2949120);
    float* spf    = alloc(512);
    float* spw    = alloc(512);
    float* spr    = alloc(512);
    float* REUSE  = alloc(8388608);           // t1/t2/t3 -> S -> fbuf/ffn2o/cat
    float* ef     = alloc(512);
    float* ew     = alloc(512);
    float* er     = alloc(512);
    float* sw     = alloc(8);                 // sinkhorn weights
    float* zs     = alloc(512);
    float* zimg   = alloc(524288);
    float* d3     = alloc(262144);
    float* d2     = alloc(262144);
    float* dfin   = alloc(1572864);           // df,dw,dr [3][524288]
    float* comb   = alloc(1572864);
    float* cdout  = alloc(524288);
    float* b2     = alloc(524288);
    float* qkvb   = alloc(1048576);
    float* obuf   = alloc(524288);
    float* pebuf  = alloc(524288);
    float* attn   = alloc(524288);
    float* b3     = alloc(524288);
    float* b4     = alloc(524288);

    float* t1 = REUSE;                 // [4][64][5760] max
    float* t2 = REUSE + 1572864;       // [4][128][5760] max
    float* t3 = REUSE + 4718592;       // [4][128][5760] max
    float* S  = REUSE;                 // [8][1024][1024]
    float* fb = REUSE;                 // ffn hidden [4][256][1024]
    float* f2 = REUSE + 1048576;       // ffn2 out   [4][128][1024]
    float* ct = REUSE + 1572864;       // concat     [4][256][1024]

    const int EW = 256; // elementwise block

    // 1) cv1: silu(bn(conv1x1(x)))  M=256 K=256 N=1024, 4 batches
    gemm<0,0>(stream, P(1), X, y, 256, 1024, 256, 256, 1024, 1024,
              0, 262144, 262144, P(2), P(3), 1, 2, 4);

    // 2) cross-domain transforms of b (= y channels 128..255)
    k_dft_rows<<<2048, EW, 0, stream>>>(y, tr, ti);
    k_dft_cols<<<2048, EW, 0, stream>>>(tr, ti, freq);
    k_pool    <<<512,  EW, 0, stream>>>(y, yl);
    k_radon   <<<(2949120 + EW - 1) / EW, EW, 0, stream>>>(y, sino);
    k_rowreduce<<<512, 256, 0, stream>>>(freq, spf, 1024, 0);
    k_rowreduce<<<512, 256, 0, stream>>>(yl,   spw, 256,  0);
    k_rowreduce<<<512, 256, 0, stream>>>(sino, spr, 5760, 0);

    // 3) encoders (implicit-GEMM WMMA convs) on freq / yl / sino
    struct Enc { const float* src; int H, Wd; float* out; };
    Enc encs[3] = { {freq, 32, 32, ef}, {yl, 16, 16, ew}, {sino, 180, 32, er} };
    for (int e = 0; e < 3; ++e) {
        int HWs = encs[e].H * encs[e].Wd;
        gemm<0,2>(stream, P(22), encs[e].src, t1, 64, HWs, 1152, 0, 0, HWs,
                  0, (long)128 * HWs, (long)64 * HWs, nullptr, P(23), 1, 1, 4,
                  encs[e].H, encs[e].Wd);
        gemm<0,2>(stream, P(24), t1, t2, 128, HWs, 576, 0, 0, HWs,
                  0, (long)64 * HWs, (long)128 * HWs, nullptr, P(25), 1, 1, 4,
                  encs[e].H, encs[e].Wd);
        gemm<0,2>(stream, P(26), t2, t3, 128, HWs, 1152, 0, 0, HWs,
                  0, (long)128 * HWs, (long)128 * HWs, nullptr, P(27), 1, 1, 4,
                  encs[e].H, encs[e].Wd);
        k_rowreduce<<<512, 256, 0, stream>>>(t3, encs[e].out, HWs, 1);
    }

    // 4) sinkhorn weights
    k_sinkhorn<<<1, 32, 0, stream>>>(ef, ew, er, sw);

    // 5) decoders
    struct Dec { const float* e; int widx; int base; float* out; };
    Dec decs[3] = { {ef, 0, 28, dfin}, {ew, 0, 36, dfin + 524288}, {er, 1, 44, dfin + 1048576} };
    for (int t = 0; t < 3; ++t) {
        int q = decs[t].base;
        k_scalevec<<<2, 256, 0, stream>>>(zs, decs[t].e, sw, decs[t].widx);
        // fc: M=4(pad), N=131072, K=128 ; bias per-col
        gemm<0,1>(stream, zs, P(q + 0), zimg, 4, 131072, 128, 128, 128, 131072,
                  0, 0, 0, nullptr, P(q + 1), 2, 0, 1);
        // tconv l3: 128->64, relu
        gemm<2,2>(stream, P(q + 2), zimg, d3, 64, 1024, 1152, 0, 0, 1024,
                  0, 131072, 65536, nullptr, P(q + 3), 1, 1, 4, 32, 32, 64);
        // tconv l2: 64->64, relu
        gemm<2,2>(stream, P(q + 4), d3, d2, 64, 1024, 576, 0, 0, 1024,
                  0, 65536, 65536, nullptr, P(q + 5), 1, 1, 4, 32, 32, 64);
        // tconv l1: 64->128, sigmoid
        gemm<2,2>(stream, P(q + 6), d2, decs[t].out, 128, 1024, 576, 0, 0, 1024,
                  0, 65536, 131072, nullptr, P(q + 7), 1, 3, 4, 32, 32, 128);
    }

    // 6) comb + c1x1 sigmoid + gating
    k_comb<<<(1572864 + EW - 1) / EW, EW, 0, stream>>>(
        y, dfin, dfin + 524288, dfin + 1048576, spf, spw, spr, comb);
    gemm<0,0>(stream, P(52), comb, cdout, 128, 1024, 384, 384, 1024, 1024,
              0, 393216, 131072, nullptr, P(53), 1, 3, 4);
    k_bmul<<<2048, EW, 0, stream>>>(y, cdout, b2);

    // 7) attention
    gemm<0,0>(stream, P(7), b2, qkvb, 256, 1024, 128, 128, 1024, 1024,
              0, 131072, 262144, P(8), P(9), 1, 0, 4);
    // S = q^T k  (8 batches of (b,h))
    gemm<1,0>(stream, qkvb, qkvb + 32768, S, 1024, 1024, 32, 1024, 1024, 1024,
              131072, 131072, 1048576, nullptr, nullptr, 0, 0, 8);
    k_softmax<<<8192, 256, 0, stream>>>(S, 0.17677669529663687f);
    // o = V P^T
    gemm<0,1>(stream, qkvb + 65536, S, obuf, 64, 1024, 1024, 1024, 1024, 1024,
              131072, 1048576, 65536, nullptr, nullptr, 0, 0, 8);
    k_dw_pe<<<2048, EW, 0, stream>>>(qkvb, P(10), P(11), P(12), pebuf);
    k_add<<<2048, EW, 0, stream>>>(obuf, obuf, pebuf, 524288);
    gemm<0,0>(stream, P(13), obuf, attn, 128, 1024, 128, 128, 1024, 1024,
              0, 131072, 131072, P(14), P(15), 1, 0, 4);
    k_add<<<2048, EW, 0, stream>>>(b3, b2, attn, 524288);

    // 8) FFN
    gemm<0,0>(stream, P(16), b3, fb, 256, 1024, 128, 128, 1024, 1024,
              0, 131072, 262144, P(17), P(18), 1, 2, 4);
    gemm<0,0>(stream, P(19), fb, f2, 128, 1024, 256, 256, 1024, 1024,
              0, 262144, 131072, P(20), P(21), 1, 0, 4);
    k_add<<<2048, EW, 0, stream>>>(b4, b3, f2, 524288);

    // 9) concat + cv2 (silu) -> output
    k_cat<<<4096, EW, 0, stream>>>(y, b4, ct);
    gemm<0,0>(stream, P(4), ct, (float*)d_out, 256, 1024, 256, 256, 1024, 1024,
              0, 262144, 262144, P(5), P(6), 1, 2, 4);
}
